// DynamicConv_5961414607038
// MI455X (gfx1250) — matrix operand order repaired and checked by reading, not compiled, not run on hardware
//
#include <hip/hip_runtime.h>
#include <math.h>

// Problem constants (from reference setup_inputs)
#define BATCH 32
#define CH    128
#define HH    128
#define WW    128
#define HWSZ  (HH*WW)          // 16384
#define NBAS  4
#define KKT   9                // 3x3 taps
#define JD    (CH*KKT)         // 1152 per-base kernel elements
// ws layout (floats): [0,4096) pooled_sum, [4096,4224) asum, [4224,41088) kern

typedef __attribute__((ext_vector_type(16))) _Float16 v16h;
typedef __attribute__((ext_vector_type(8)))  float    v8f;
typedef __attribute__((ext_vector_type(4)))  unsigned int v4u;
typedef __attribute__((ext_vector_type(8)))  int          v8i;
typedef __attribute__((ext_vector_type(4)))  int          v4i;

// ---------------------------------------------------------------- kernel 0
__global__ void k_init(float* __restrict__ ws) {
    int i = blockIdx.x * 256 + threadIdx.x;
    if (i < 4224) ws[i] = 0.0f;
}

// ---------------------------------------------------------------- kernel A
// Fused pass 1 over x: pooled channel sums + per-pixel attn softmax sums.
// grid = (8 chunks of 2048 px, 32 batches), block = 256 (8 wave32).
__global__ __launch_bounds__(256) void k_stats(
        const float* __restrict__ x, const float* __restrict__ w_attn,
        const float* __restrict__ b_attn,
        float* __restrict__ pooled, float* __restrict__ asum) {
    __shared__ float lwa[NBAS * CH];
    const int tid = threadIdx.x;
    for (int i = tid; i < NBAS * CH; i += 256) lwa[i] = w_attn[i];
    __syncthreads();

    const int b = blockIdx.y;
    const int pixbase = blockIdx.x * 2048;        // this block's pixel chunk
    const float4* x4 = reinterpret_cast<const float4*>(x);

    float lg[8][4];
    #pragma unroll
    for (int p = 0; p < 8; ++p)
        #pragma unroll
        for (int n = 0; n < 4; ++n) lg[p][n] = 0.0f;

    for (int c = 0; c < CH; ++c) {
        size_t base4 = (((size_t)(b * CH + c) * HWSZ) + pixbase) >> 2;
        float4 va = x4[base4 + tid * 2];
        float4 vb = x4[base4 + tid * 2 + 1];
        float vv[8] = {va.x, va.y, va.z, va.w, vb.x, vb.y, vb.z, vb.w};
        float w0 = lwa[c], w1 = lwa[CH + c], w2 = lwa[2 * CH + c], w3 = lwa[3 * CH + c];
        float s = 0.0f;
        #pragma unroll
        for (int p = 0; p < 8; ++p) {
            s += vv[p];
            lg[p][0] = fmaf(vv[p], w0, lg[p][0]);
            lg[p][1] = fmaf(vv[p], w1, lg[p][1]);
            lg[p][2] = fmaf(vv[p], w2, lg[p][2]);
            lg[p][3] = fmaf(vv[p], w3, lg[p][3]);
        }
        // wave32 reduction for pooled channel sum
        #pragma unroll
        for (int off = 16; off > 0; off >>= 1) s += __shfl_xor(s, off, 32);
        if ((tid & 31) == 0) atomicAdd(&pooled[b * CH + c], s);
    }

    // per-pixel softmax over the 4 bases, accumulate spatial sum of attn
    float ba0 = b_attn[0], ba1 = b_attn[1], ba2 = b_attn[2], ba3 = b_attn[3];
    float at[4] = {0.f, 0.f, 0.f, 0.f};
    #pragma unroll
    for (int p = 0; p < 8; ++p) {
        float l0 = lg[p][0] + ba0, l1 = lg[p][1] + ba1;
        float l2 = lg[p][2] + ba2, l3 = lg[p][3] + ba3;
        float m = fmaxf(fmaxf(l0, l1), fmaxf(l2, l3));
        float e0 = __expf(l0 - m), e1 = __expf(l1 - m);
        float e2 = __expf(l2 - m), e3 = __expf(l3 - m);
        float inv = 1.0f / (e0 + e1 + e2 + e3);
        at[0] += e0 * inv; at[1] += e1 * inv; at[2] += e2 * inv; at[3] += e3 * inv;
    }
    #pragma unroll
    for (int n = 0; n < 4; ++n) {
        float v = at[n];
        #pragma unroll
        for (int off = 16; off > 0; off >>= 1) v += __shfl_xor(v, off, 32);
        if ((tid & 31) == 0) atomicAdd(&asum[b * 4 + n], v);
    }
}

// ---------------------------------------------------------------- kernel B
// bases GEMM via WMMA: bases[b, j] = pooled_mean[b,:] . w_base[j,:],
// fused with attention weighting + bias -> kern[b, c*9+tap].
// One wave per (mt in 0..1, jt in 0..71) 16x16 output tile; K=128 as 4x
// v_wmma_f32_16x16x32_f16; loop over 4 bases (n) reusing the A fragments.
__global__ __launch_bounds__(32) void k_kern(
        const float* __restrict__ pooled, const float* __restrict__ asum,
        const float* __restrict__ w_base, const float* __restrict__ b_base,
        float* __restrict__ kern) {
    const float inv_hw = 1.0f / (float)HWSZ;
    const int lane = threadIdx.x;
    const int jt = blockIdx.x;        // 72 tiles over j' = c*9+tap (1152/16)
    const int mt = blockIdx.y;        // 2 tiles over b (32/16)
    const int row = lane & 15;
    const int hi  = (lane >> 4) & 1;  // lane group 16..31
    const int col = lane & 15;

    // A fragments (16x32 f16, lane = row M, per ISA layout):
    // elems 0..7  -> K = k0 + (hi?8:0) + e ; elems 8..15 -> +16
    v16h afrag[4];
    #pragma unroll
    for (int kt = 0; kt < 4; ++kt) {
        const int k0 = kt * 32;
        #pragma unroll
        for (int e = 0; e < 16; ++e) {
            int k = k0 + ((e & 8) ? 16 : 0) + (hi ? 8 : 0) + (e & 7);
            afrag[kt][e] = (_Float16)(pooled[(mt * 16 + row) * CH + k] * inv_hw);
        }
    }

    float ksum[8];
    #pragma unroll
    for (int r = 0; r < 8; ++r) ksum[r] = 0.0f;

    #pragma unroll
    for (int n = 0; n < NBAS; ++n) {
        const int j = n * JD + jt * 16 + col;   // B column this lane owns
        v8f acc = {0.f, 0.f, 0.f, 0.f, 0.f, 0.f, 0.f, 0.f};
        #pragma unroll
        for (int kt = 0; kt < 4; ++kt) {
            // B fragment (32x16 f16, lane = col N): elems e -> K = kb + (e&7) (+16 for e>=8)
            const int kb = kt * 32 + (hi ? 8 : 0);
            v16h bfrag;
            #pragma unroll
            for (int e = 0; e < 16; ++e)
                bfrag[e] = (_Float16)w_base[(size_t)j * CH + kb + ((e & 8) ? 16 : 0) + (e & 7)];
            acc = __builtin_amdgcn_wmma_f32_16x16x32_f16(
                    false, afrag[kt], false, bfrag, (short)0, acc, false, false);
        }
        const float bias = b_base[j];
        #pragma unroll
        for (int r = 0; r < 8; ++r) {
            int brow = mt * 16 + r + (hi ? 8 : 0);   // D: VGPR r -> M = r (+8)
            float a = asum[brow * 4 + n] * inv_hw;
            ksum[r] += a * (acc[r] + bias);
        }
    }
    #pragma unroll
    for (int r = 0; r < 8; ++r) {
        int brow = mt * 16 + r + (hi ? 8 : 0);
        kern[brow * JD + jt * 16 + col] = ksum[r];
    }
}

// ---------------------------------------------------------------- kernel C
// Pass 2: per-sample channel-summed 3x3 conv (SAME, cross-correlation).
// Tiles are staged global->LDS by the Tensor Data Mover (double-buffered,
// wave 0 issues, TENSORcnt-tracked). TDM LDS padding (128 DWORDs data,
// 8 DWORDs pad) recreates the 136-float padded row layout; halo columns
// live inside the pad gaps so the pre-zeroed halos survive every transfer.
// tensor_dim1 = rows-remaining gives hardware zero-fill of the bottom halo;
// the top halo (stripe 0) shifts lds_addr one row over a pre-zeroed row 0.
__global__ __launch_bounds__(256) void k_conv(
        const float* __restrict__ x, const float* __restrict__ kern,
        float* __restrict__ out) {
    __shared__ float lk[JD];
    __shared__ __align__(16) float lx[2][18 * 136];
    const int tid = threadIdx.x;
    const int b = blockIdx.y;
    const int rowbase = blockIdx.x * 16;
    const bool top = (rowbase == 0);

    for (int i = tid; i < JD; i += 256) lk[i] = kern[b * JD + i];
    if (tid < 36) {                  // zero left/right halo cols (pad gaps)
        int bb = tid / 18, rr = tid % 18;
        lx[bb][rr * 136 + 3] = 0.0f;
        lx[bb][rr * 136 + 132] = 0.0f;
    }
    if (top) {                       // zero top-halo row 0 of both buffers
        for (int i = tid; i < 2 * 136; i += 256) lx[i / 136][i % 136] = 0.0f;
    }
    __syncthreads();

    // --- TDM descriptor pieces (uniform per block) ---
    const int gr0   = top ? 0 : (rowbase - 1);   // first global row fetched
    const unsigned td1   = (unsigned)(HH - gr0); // rows remaining -> OOB zero
    const unsigned tile1 = top ? 17u : 18u;      // rows transferred
    const unsigned ldsoff[2] = {
        (unsigned)(uintptr_t)(void*)&lx[0][0] + (top ? 544u : 0u),
        (unsigned)(uintptr_t)(void*)&lx[1][0] + (top ? 544u : 0u)};
    // group1 word0: data_size=4B | pad_enable | pad_interval=128dw | pad=8dw
    const unsigned g1w0 = (2u << 16) | (1u << 20) | (6u << 22) | (7u << 25);

    auto tdm_issue = [&](int c, int buf) {
        size_t gaddr = (size_t)(uintptr_t)x +
                       4ull * (((size_t)(b * CH + c)) * HWSZ + (size_t)gr0 * WW);
        v4u g0 = {1u,                         // count=1, user descriptor
                  ldsoff[buf],                // lds_addr (bytes)
                  (unsigned)gaddr,            // global_addr[31:0]
                  ((unsigned)(gaddr >> 32) & 0x1FFFFFFu) | (2u << 30)}; // +type=2
        v8i g1 = {(int)g1w0,
                  (int)(128u << 16),          // tensor_dim0 = 128 (low16)
                  (int)(td1 << 16),           // dim0 hi=0 | tensor_dim1 low16
                  (int)(128u << 16),          // dim1 hi=0 | tile_dim0 = 128
                  (int)tile1,                 // tile_dim1 | tile_dim2=0
                  128,                        // tensor_dim0_stride = W
                  (int)((unsigned)HWSZ << 16),// tensor_dim1_stride low16
                  0};                         // tensor_dim1_stride hi
        v4i gz4 = {0, 0, 0, 0};               // D# groups 2/3 unused (2D tile)
        v8i gz8 = {0, 0, 0, 0, 0, 0, 0, 0};   // extra group (clang-23 form)
        __builtin_amdgcn_tensor_load_to_lds(g0, g1, gz4, gz4, gz8, 0);
    };

    if (tid < 32) tdm_issue(0, 0);   // prefetch first channel

    const int r  = tid >> 4;          // 0..15 output row in stripe
    const int cs = (tid & 15) * 8;    // first of 8 output cols
    float acc[8];
    #pragma unroll
    for (int p = 0; p < 8; ++p) acc[p] = 0.0f;

    for (int c = 0; c < CH; ++c) {
        if (tid < 32) {               // wave 0: prefetch next, wait current
            if (c + 1 < CH) {
                tdm_issue(c + 1, (c + 1) & 1);
                __builtin_amdgcn_s_wait_tensorcnt(1);  // TDM(c) done (in-order)
            } else {
                __builtin_amdgcn_s_wait_tensorcnt(0);
            }
        }
        __syncthreads();              // tile c visible to all waves

        const float* lxb = &lx[c & 1][0];
        #pragma unroll
        for (int dy = 0; dy < 3; ++dy) {
            const float* rowp = &lxb[(r + dy) * 136 + cs + 3]; // cols cs-1..cs+8
            float w0 = lk[c * 9 + dy * 3 + 0];
            float w1 = lk[c * 9 + dy * 3 + 1];
            float w2 = lk[c * 9 + dy * 3 + 2];
            float  L  = rowp[0];
            float4 A  = *reinterpret_cast<const float4*>(rowp + 1); // aligned
            float4 Bv = *reinterpret_cast<const float4*>(rowp + 5); // aligned
            float  R  = rowp[9];
            float v[10] = {L, A.x, A.y, A.z, A.w, Bv.x, Bv.y, Bv.z, Bv.w, R};
            #pragma unroll
            for (int p = 0; p < 8; ++p)
                acc[p] = fmaf(v[p], w0, fmaf(v[p + 1], w1, fmaf(v[p + 2], w2, acc[p])));
        }
        __syncthreads();              // compute done: buffer may be rewritten
    }

    size_t ob = (size_t)b * HWSZ + (size_t)(rowbase + r) * WW + cs;
    *reinterpret_cast<float4*>(&out[ob])     = make_float4(acc[0], acc[1], acc[2], acc[3]);
    *reinterpret_cast<float4*>(&out[ob + 4]) = make_float4(acc[4], acc[5], acc[6], acc[7]);
}

// ---------------------------------------------------------------- launch
extern "C" void kernel_launch(void* const* d_in, const int* in_sizes, int n_in,
                              void* d_out, int out_size, void* d_ws, size_t ws_size,
                              hipStream_t stream) {
    const float* x      = (const float*)d_in[0];
    const float* w_base = (const float*)d_in[1];
    const float* b_base = (const float*)d_in[2];
    const float* w_attn = (const float*)d_in[3];
    const float* b_attn = (const float*)d_in[4];
    float* out = (float*)d_out;
    float* ws  = (float*)d_ws;        // needs 41088 floats (~161 KB)
    float* pooled = ws;               // [32*128] channel sums
    float* asum   = ws + 4096;        // [32*4]   attention sums
    float* kern   = ws + 4224;        // [32*1152] dynamic kernels

    k_init <<<dim3(17),    dim3(256), 0, stream>>>(ws);
    k_stats<<<dim3(8, 32), dim3(256), 0, stream>>>(x, w_attn, b_attn, pooled, asum);
    k_kern <<<dim3(72, 2), dim3(32),  0, stream>>>(pooled, asum, w_base, b_base, kern);
    k_conv <<<dim3(8, 32), dim3(256), 0, stream>>>(x, kern, out);
}
